// DeformableConvModule_12043088298164
// MI455X (gfx1250) — matrix operand order repaired and checked by reading, not compiled, not run on hardware
//
#include <hip/hip_runtime.h>

typedef __attribute__((ext_vector_type(16))) _Float16 v16h;
typedef __attribute__((ext_vector_type(8)))  _Float16 v8h;
typedef __attribute__((ext_vector_type(8)))  float    v8f;
typedef __attribute__((ext_vector_type(4)))  float    v4f;

#define NB    16
#define CIN   64
#define COUT  64
#define HIN   66
#define WIN   66
#define HOUT  64
#define WOUT  64
#define KTAPS 9
#define KTOT  576          // CIN * KTAPS
#define KSTEPS 18          // KTOT / 32
#define ROWPAD 584         // 576 + 8 halfs pad (keeps 16B alignment, breaks bank pattern)
#define MROWS  32          // output pixels per block

// ---------------------------------------------------------------------------
// Weight prep: f32 weight[cout][cin][3][3] -> fragment-ready f16 in workspace.
// Layout: wf[((ntile*18 + ks)*32 + lane)*16 + j]
//   B-matrix 32x16 f16 layout: lanes 0-15 hold K=ks*32+0..7,16..23 (j=0..15),
//   lanes 16-31 hold K=ks*32+8..15,24..31; column n = lane&15.
//   kk = k*64 + c  (tap-major, channel-minor) matches the LDS im2col layout.
// ---------------------------------------------------------------------------
__global__ void __launch_bounds__(256)
wprep(const float* __restrict__ w, _Float16* __restrict__ wf) {
  int idx = blockIdx.x * 256 + threadIdx.x;        // 4*18*32*16 = 36864 total
  if (idx >= 4 * KSTEPS * 32 * 16) return;
  int j    = idx & 15;
  int lane = (idx >> 4) & 31;
  int ks   = (idx >> 9) % KSTEPS;
  int nt   = (idx >> 9) / KSTEPS;
  int g = lane >> 4, r = lane & 15;
  int kk = ks * 32 + (j >> 3) * 16 + g * 8 + (j & 7);
  int k  = kk >> 6;          // tap 0..8
  int c  = kk & 63;          // input channel
  int o  = nt * 16 + r;      // output channel
  wf[idx] = (_Float16)w[(o * CIN + c) * KTAPS + k];
}

// ---------------------------------------------------------------------------
// Fused deformable-im2col (global gather -> LDS f16) + WMMA GEMM.
// One block = one (n, ho, half-row of 32 wo).
// ---------------------------------------------------------------------------
__global__ void __launch_bounds__(256)
deform_wmma(const float* __restrict__ inp, const float* __restrict__ off,
            const _Float16* __restrict__ wf, float* __restrict__ out) {
  __shared__ _Float16 sval[MROWS * ROWPAD];        // 37376 bytes

  const int tid = threadIdx.x;
  const int bid = blockIdx.x;                      // N * HOUT * 2 = 2048
  const int n   = bid >> 7;
  const int ho  = (bid >> 1) & 63;
  const int wob = (bid & 1) * MROWS;

  // ---- Phase A: bilinear sampling into LDS ----
  const float* ibase = inp + n * (CIN * HIN * WIN);
  for (int t = tid; t < KTAPS * MROWS; t += 256) {
    const int k   = t >> 5;                        // tap 0..8
    const int wol = t & 31;
    const int wo  = wob + wol;
    const int ob  = ((n * (2 * KTAPS) + 2 * k) * HOUT + ho) * WOUT + wo;
    const float dy = off[ob];
    const float dx = off[ob + HOUT * WOUT];
    const float py = (float)(ho + (k / 3)) + dy;
    const float px = (float)(wo + (k % 3)) + dx;
    const float y0f = floorf(py), x0f = floorf(px);
    const float fy = py - y0f,  fx = px - x0f;
    const int y0 = (int)y0f, x0 = (int)x0f;
    const int y1 = y0 + 1,   x1 = x0 + 1;
    const float vy0 = (y0 >= 0 && y0 < HIN) ? 1.f : 0.f;
    const float vy1 = (y1 >= 0 && y1 < HIN) ? 1.f : 0.f;
    const float vx0 = (x0 >= 0 && x0 < WIN) ? 1.f : 0.f;
    const float vx1 = (x1 >= 0 && x1 < WIN) ? 1.f : 0.f;
    const float w00 = (1.f - fy) * (1.f - fx) * vy0 * vx0;
    const float w01 = (1.f - fy) * fx         * vy0 * vx1;
    const float w10 = fy         * (1.f - fx) * vy1 * vx0;
    const float w11 = fy         * fx         * vy1 * vx1;
    const int y0c = min(max(y0, 0), HIN - 1), y1c = min(max(y1, 0), HIN - 1);
    const int x0c = min(max(x0, 0), WIN - 1), x1c = min(max(x1, 0), WIN - 1);
    const int i00 = y0c * WIN + x0c;
    const int i01 = y0c * WIN + x1c;
    const int i10 = y1c * WIN + x0c;
    const int i11 = y1c * WIN + x1c;
    _Float16* dst = &sval[wol * ROWPAD + k * CIN];
    const float* p = ibase;
    #pragma unroll 4
    for (int c = 0; c < CIN; ++c) {
      const float v = w00 * p[i00] + w01 * p[i01] + w10 * p[i10] + w11 * p[i11];
      dst[c] = (_Float16)v;
      p += HIN * WIN;
    }
  }
  __syncthreads();

  // ---- Phase B: GEMM via v_wmma_f32_16x16x32_f16 ----
  const int wave = tid >> 5, lane = tid & 31;
  const int r = lane & 15, g = lane >> 4;
  const int mtile = wave >> 2;                     // 0..1
  const int ntile = wave & 3;                      // 0..3

  // A 16x32 f16 layout: lane r holds row m=r, halfs j: K = (j>>3)*16 + g*8 + (j&7)
  const _Float16* abase = &sval[(mtile * 16 + r) * ROWPAD + g * 8];
  const _Float16* bbase = wf + (ntile * KSTEPS) * 32 * 16 + lane * 16;

  v8f acc = {};
  #pragma unroll
  for (int ks = 0; ks < KSTEPS; ++ks) {
    const v8h alo = *(const v8h*)(abase + ks * 32);        // K = base + g*8 + 0..7
    const v8h ahi = *(const v8h*)(abase + ks * 32 + 16);   // K = base+16 + g*8 + 0..7
    const v16h a = __builtin_shufflevector(alo, ahi,
        0, 1, 2, 3, 4, 5, 6, 7, 8, 9, 10, 11, 12, 13, 14, 15);
    const v16h b = *(const v16h*)(bbase + ks * 32 * 16);
    acc = __builtin_amdgcn_wmma_f32_16x16x32_f16(
        /*neg_a=*/false, a, /*neg_b=*/false, b,
        /*c_mod=*/(short)0, acc, /*reuse_a=*/false, /*reuse_b=*/false);
  }

  // D layout: lane (r,g), VGPR i -> pixel m = 8g + i, cout = r
  const int cout = ntile * 16 + r;
  const int wo0  = wob + mtile * 16 + g * 8;
  float* op = out + ((n * COUT + cout) * HOUT + ho) * WOUT + wo0;
  const v4f lo = { acc[0], acc[1], acc[2], acc[3] };
  const v4f hi = { acc[4], acc[5], acc[6], acc[7] };
  *(v4f*)op       = lo;
  *(v4f*)(op + 4) = hi;
}

extern "C" void kernel_launch(void* const* d_in, const int* in_sizes, int n_in,
                              void* d_out, int out_size, void* d_ws, size_t ws_size,
                              hipStream_t stream) {
  (void)in_sizes; (void)n_in; (void)out_size; (void)ws_size;
  const float* inp = (const float*)d_in[0];   // [16,64,66,66] f32
  const float* off = (const float*)d_in[1];   // [16,18,64,64] f32
  const float* wgt = (const float*)d_in[2];   // [64,64,3,3]   f32
  float* out = (float*)d_out;                 // [16,64,64,64] f32
  _Float16* wfrag = (_Float16*)d_ws;          // 73728 bytes of fragment-ready f16 weights

  wprep<<<144, 256, 0, stream>>>(wgt, wfrag);
  deform_wmma<<<NB * HOUT * 2, 256, 0, stream>>>(inp, off, wfrag, out);
}
